// Lupool_13804024889769
// MI455X (gfx1250) — matrix operand order repaired and checked by reading, not compiled, not run on hardware
//
#include <hip/hip_runtime.h>
#include <hip/hip_bf16.h>
#include <math.h>

typedef __attribute__((ext_vector_type(16))) _Float16 v16h;
typedef __attribute__((ext_vector_type(8)))  float    v8f;

#define TPB 256
#define NEG_SLOPE 0.2f

// ---------------------------------------------------------------------------
// Operand pre-swizzle into WMMA lane layouts (CDNA5 ISA 7.12.2, wave32).
// Packed form: [tile][kstep][lane 0..31][16 f16]  (each lane's 16 elems are a
// contiguous 32B run -> the GEMM loads one v16h per operand per k-step).
//
// A (16x32 f16): lane l (half=l>>4, m=l&15): elem j holds K = half*8 + j + (j&8)
// B (32x16 f16): lane l (n=l&15):            elem j holds K = half*8 + j + (j&8)
// C/D (16x16 f32): lane l: c[v] = (M = v + half*8, N = l&15)
// ---------------------------------------------------------------------------
__global__ void pack_a_k(const float* __restrict__ X, _Float16* __restrict__ Ap,
                         int M, int K, int total) {
  int idx = blockIdx.x * blockDim.x + threadIdx.x;   // rowTiles*KS*512
  if (idx >= total) return;
  const int KS   = K >> 5;
  const int j    = idx & 15;
  const int lane = (idx >> 4) & 31;
  const int t    = idx >> 9;            // rt*KS + ks
  const int ks   = t % KS;
  const int rt   = t / KS;
  const int row  = min((rt << 4) + (lane & 15), M - 1);
  const int kk   = (ks << 5) + ((lane >> 4) << 3) + j + (j & 8);
  Ap[idx] = (_Float16)X[row * K + kk];
}

__global__ void pack_b_k(const float* __restrict__ W, _Float16* __restrict__ Wp,
                         int K, int OUTD, int total) {
  int idx = blockIdx.x * blockDim.x + threadIdx.x;   // (OUTD/16)*KS*512
  if (idx >= total) return;
  const int KS   = K >> 5;
  const int j    = idx & 15;
  const int lane = (idx >> 4) & 31;
  const int t    = idx >> 9;            // ct*KS + ks
  const int ks   = t % KS;
  const int ct   = t / KS;
  const int kk   = (ks << 5) + ((lane >> 4) << 3) + j + (j & 8);
  const int col  = (ct << 4) + (lane & 15);
  Wp[idx] = (_Float16)W[kk * OUTD + col];
}

// ---------------------------------------------------------------------------
// WMMA GEMM with bias: OUT[M,OUTD] = X[M,K] @ W[K,OUTD] + b  (operands pre-
// swizzled f16, f32 accumulate). One wave per 16x16 D tile; a block holds
// OUTD/16 waves per 16-row stripe. Inner loop: 2 v16h loads + 1 WMMA.
// ---------------------------------------------------------------------------
template <int K>
__global__ void wmma_gemm_bias_h(const _Float16* __restrict__ Ap,
                                 const _Float16* __restrict__ Wp,
                                 const float* __restrict__ b,
                                 float* __restrict__ OUT,
                                 int M, int OUTD) {
  constexpr int KS = K / 32;
  const int lane = threadIdx.x & 31;
  const int wave = threadIdx.x >> 5;   // column tile
  const int row0 = blockIdx.x << 4;
  const int col0 = wave << 4;
  const int half = lane >> 4;
  const int mn   = lane & 15;

  v8f c;
  const float bias = b[col0 + mn];
#pragma unroll
  for (int v = 0; v < 8; ++v) c[v] = bias;

  const v16h* __restrict__ ap = (const v16h*)Ap + (blockIdx.x * KS) * 32 + lane;
  const v16h* __restrict__ wp = (const v16h*)Wp + (wave * KS) * 32 + lane;

#pragma unroll
  for (int ks = 0; ks < KS; ++ks) {
    const v16h av = ap[ks * 32];
    const v16h bv = wp[ks * 32];
    c = __builtin_amdgcn_wmma_f32_16x16x32_f16(false, av, false, bv,
                                               (short)0, c, false, false);
  }

  if (row0 + 16 <= M) {  // wave-uniform fast path: straight-line stores
#pragma unroll
    for (int v = 0; v < 8; ++v)
      OUT[(row0 + v + (half << 3)) * OUTD + col0 + mn] = c[v];
  } else {
#pragma unroll
    for (int v = 0; v < 8; ++v) {
      const int m = row0 + v + (half << 3);
      if (m < M) OUT[m * OUTD + col0 + mn] = c[v];
    }
  }
}

// ------------------------- ordered-float atomic max -------------------------
__device__ __forceinline__ unsigned f2ord(float x) {
  unsigned u = __float_as_uint(x);
  return (u >> 31) ? ~u : (u | 0x80000000u);
}
__device__ __forceinline__ float ord2f(unsigned o) {
  unsigned u = (o >> 31) ? (o & 0x7fffffffu) : ~o;
  return __uint_as_float(u);
}

// e[eh] = sum_d leakyrelu(fs[src][h,d] + fd[dst][h,d]) * attn[h,d]; seg-max to mmax
// D % 4 == 0; all row bases are 16B aligned -> float4 loads.
__global__ void edge_scores_k(const float* __restrict__ fs, const float* __restrict__ fd,
                              const int* __restrict__ src, const int* __restrict__ dst,
                              const float* __restrict__ attn,
                              float* __restrict__ e, unsigned* __restrict__ mmax,
                              int E, int H, int D) {
  long long idx = (long long)blockIdx.x * blockDim.x + threadIdx.x;
  if (idx >= (long long)E * H) return;
  const int h  = (int)(idx % H);
  const int ei = (int)(idx / H);
  const float4* ps = (const float4*)(fs + (long long)src[ei] * H * D + (long long)h * D);
  const float4* pd = (const float4*)(fd + (long long)dst[ei] * H * D + (long long)h * D);
  const float4* pa = (const float4*)(attn + (long long)h * D);
  float acc = 0.f;
  for (int d = 0; d < (D >> 2); ++d) {
    const float4 a = ps[d], bq = pd[d], w = pa[d];
    float f0 = a.x + bq.x, f1 = a.y + bq.y, f2 = a.z + bq.z, f3 = a.w + bq.w;
    f0 = f0 > 0.f ? f0 : NEG_SLOPE * f0;
    f1 = f1 > 0.f ? f1 : NEG_SLOPE * f1;
    f2 = f2 > 0.f ? f2 : NEG_SLOPE * f2;
    f3 = f3 > 0.f ? f3 : NEG_SLOPE * f3;
    acc += f0 * w.x + f1 * w.y + f2 * w.z + f3 * w.w;
  }
  e[(long long)ei * H + h] = acc;
  atomicMax(&mmax[(long long)dst[ei] * H + h], f2ord(acc));
}

// m[i] = finite(max) ? max : 0   (0-ord == "never touched")
__global__ void finalize_max_k(const unsigned* __restrict__ mm, float* __restrict__ m, int n) {
  int i = blockIdx.x * blockDim.x + threadIdx.x;
  if (i >= n) return;
  unsigned o = mm[i];
  float v = (o == 0u) ? 0.f : ord2f(o);
  if (!isfinite(v)) v = 0.f;
  m[i] = v;
}

// in place: e -> exp(e - m[dst]); seg-sum to s
__global__ void edge_exp_k(float* __restrict__ ea, const float* __restrict__ m,
                           const int* __restrict__ dst, float* __restrict__ s,
                           int E, int H) {
  long long idx = (long long)blockIdx.x * blockDim.x + threadIdx.x;
  if (idx >= (long long)E * H) return;
  const int h  = (int)(idx % H);
  const int ei = (int)(idx / H);
  float a = __expf(ea[idx] - m[(long long)dst[ei] * H + h]);
  ea[idx] = a;
  atomicAdd(&s[(long long)dst[ei] * H + h], a);
}

// in place: a -> a / s[dst]
__global__ void alpha_div_k(float* __restrict__ ea, const float* __restrict__ s,
                            const int* __restrict__ dst, int E, int H) {
  long long idx = (long long)blockIdx.x * blockDim.x + threadIdx.x;
  if (idx >= (long long)E * H) return;
  const int h  = (int)(idx % H);
  const int ei = (int)(idx / H);
  ea[idx] /= s[(long long)dst[ei] * H + h];
}

// out[dst, hd] += alpha[e,h] * fs[src, hd]
__global__ void edge_aggregate_k(const float* __restrict__ fs, const float* __restrict__ alpha,
                                 const int* __restrict__ src, const int* __restrict__ dst,
                                 float* __restrict__ out, int E, int H, int D) {
  long long idx = (long long)blockIdx.x * blockDim.x + threadIdx.x;
  const long long tot = (long long)E * H * D;
  if (idx >= tot) return;
  const int hd = (int)(idx % (H * D));
  const int ei = (int)(idx / (H * D));
  const int h  = hd / D;
  const float al = alpha[(long long)ei * H + h];
  atomicAdd(&out[(long long)dst[ei] * H * D + hd],
            al * fs[(long long)src[ei] * H * D + hd]);
}

__global__ void relu_k(float* __restrict__ p, long long n) {
  long long i = (long long)blockIdx.x * blockDim.x + threadIdx.x;
  if (i < n) p[i] = fmaxf(p[i], 0.f);
}

__global__ void degrees_k(const int* __restrict__ src, const int* __restrict__ dst,
                          float* __restrict__ degO, float* __restrict__ degI, int E) {
  int i = blockIdx.x * blockDim.x + threadIdx.x;
  if (i >= E) return;
  atomicAdd(&degO[src[i]], 1.f);
  atomicAdd(&degI[dst[i]], 1.f);
}

// h[n] = (g2[n,:] . Wg) * rsqrt(max(deg_out,1))
__global__ void hproj_k(const float* __restrict__ g2, const float* __restrict__ Wg,
                        const float* __restrict__ degO, float* __restrict__ hb,
                        int N, int D) {
  int n = blockIdx.x * blockDim.x + threadIdx.x;
  if (n >= N) return;
  float acc = 0.f;
  const float* row = g2 + (long long)n * D;
  for (int d = 0; d < D; ++d) acc += row[d] * Wg[d];
  hb[n] = acc * rsqrtf(fmaxf(degO[n], 1.f));
}

__global__ void hagg_k(const float* __restrict__ hb, const int* __restrict__ src,
                       const int* __restrict__ dst, float* __restrict__ agg, int E) {
  int i = blockIdx.x * blockDim.x + threadIdx.x;
  if (i >= E) return;
  atomicAdd(&agg[dst[i]], hb[src[i]]);
}

__global__ void scores_k(const float* __restrict__ agg, const float* __restrict__ degI,
                         const float* __restrict__ bg, float* __restrict__ sc, int N) {
  int n = blockIdx.x * blockDim.x + threadIdx.x;
  if (n >= N) return;
  sc[n] = agg[n] * rsqrtf(fmaxf(degI[n], 1.f)) + bg[0];
}

// Exact stable top-k by rank counting (ties: lower index wins, like lax.top_k).
// Every rank < k is produced by exactly one node -> d_out fully written.
__global__ void topk_write_k(const float* __restrict__ scores, const float* __restrict__ g2,
                             float* __restrict__ out, int N, int k, int D) {
  __shared__ float sh[TPB];
  const int i = blockIdx.x * blockDim.x + threadIdx.x;
  const float si = (i < N) ? scores[i] : -INFINITY;
  int rank = 0;
  for (int base = 0; base < N; base += TPB) {
    const int j = base + threadIdx.x;
    sh[threadIdx.x] = (j < N) ? scores[j] : -INFINITY;
    __syncthreads();
    const int lim = min(TPB, N - base);
    for (int t = 0; t < lim; ++t) {
      const float sj = sh[t];
      rank += (sj > si) || (sj == si && (base + t) < i);
    }
    __syncthreads();
  }
  if (i < N && rank < k) {
    const float g = tanhf(si);
    const float* row = g2 + (long long)i * D;
    float* orow = out + (long long)rank * D;
    for (int d = 0; d < D; ++d) orow[d] = row[d] * g;
  }
}

static inline unsigned nblk(long long n, int t) { return (unsigned)((n + t - 1) / t); }

extern "C" void kernel_launch(void* const* d_in, const int* in_sizes, int n_in,
                              void* d_out, int out_size, void* d_ws, size_t ws_size,
                              hipStream_t stream) {
  const float* x     = (const float*)d_in[0];
  const int*   src   = (const int*)d_in[1];
  const int*   dst   = (const int*)d_in[2];
  const float* W1s   = (const float*)d_in[3];
  const float* b1s   = (const float*)d_in[4];
  const float* W1d   = (const float*)d_in[5];
  const float* b1d   = (const float*)d_in[6];
  const float* attn1 = (const float*)d_in[7];
  const float* W2s   = (const float*)d_in[8];
  const float* b2s   = (const float*)d_in[9];
  const float* W2d   = (const float*)d_in[10];
  const float* b2d   = (const float*)d_in[11];
  const float* attn2 = (const float*)d_in[12];
  const float* Wg    = (const float*)d_in[13];
  const float* bg    = (const float*)d_in[14];

  const int IN  = 128;
  const int HID = 64;
  const int N = in_sizes[0] / IN;
  const int E = in_sizes[1];
  const int k = out_size / HID;   // = int(N*0.8)

  const int rowTiles = (N + 15) / 16;
  const int KS = IN / 32;                       // 4 k-steps
  const int apElems  = rowTiles * KS * 512;     // packed activation elems
  const int wpElems1 = (IN / 16) * KS * 512;    // packed 128x128 weight
  const int wpElems2 = (HID / 16) * KS * 512;   // packed 128x64 weight

  // ---- workspace layout (aliased; ~90 MB total) ----
  char* ws = (char*)d_ws;
  size_t off = 0;
  auto alloc = [&](size_t bytes) -> char* {
    char* p = ws + off;
    off = (off + bytes + 255) & ~(size_t)255;
    return p;
  };
  float*     A    = (float*)alloc((size_t)N * IN * 4);      // fs1; later fs2|fd2
  float*     B    = (float*)alloc((size_t)N * IN * 4);      // fd1 -> g1 -> g2
  float*     C    = (float*)alloc((size_t)E * 2 * 4);       // e -> alpha (both layers)
  _Float16*  xp   = (_Float16*)alloc((size_t)apElems * 2);  // packed x (A-operand)
  _Float16*  g1p  = (_Float16*)alloc((size_t)apElems * 2);  // packed g1 (A-operand)
  _Float16*  w1sp = (_Float16*)alloc((size_t)wpElems1 * 2);
  _Float16*  w1dp = (_Float16*)alloc((size_t)wpElems1 * 2);
  _Float16*  w2sp = (_Float16*)alloc((size_t)wpElems2 * 2);
  _Float16*  w2dp = (_Float16*)alloc((size_t)wpElems2 * 2);
  unsigned*  mm   = (unsigned*)alloc((size_t)N * 2 * 4);
  float*     m    = (float*)alloc((size_t)N * 2 * 4);
  float*     s    = (float*)alloc((size_t)N * 2 * 4);
  float*     degO = (float*)alloc((size_t)N * 4);
  float*     degI = (float*)alloc((size_t)N * 4);
  float*     hb   = (float*)alloc((size_t)N * 4);
  float*     agg  = (float*)alloc((size_t)N * 4);
  float*     sc   = (float*)alloc((size_t)N * 4);
  (void)ws_size; (void)n_in;

  float* fs2 = A;                         // [N,64]
  float* fd2 = A + (size_t)N * HID;       // [N,64]
  float* g2  = B;                         // [N,64] (after layer 2)

  // ===== pre-swizzle operands into WMMA layouts (f16) =====
  pack_a_k<<<nblk(apElems, TPB), TPB, 0, stream>>>(x, xp, N, IN, apElems);
  pack_b_k<<<nblk(wpElems1, TPB), TPB, 0, stream>>>(W1s, w1sp, IN, IN, wpElems1);
  pack_b_k<<<nblk(wpElems1, TPB), TPB, 0, stream>>>(W1d, w1dp, IN, IN, wpElems1);
  pack_b_k<<<nblk(wpElems2, TPB), TPB, 0, stream>>>(W2s, w2sp, IN, HID, wpElems2);
  pack_b_k<<<nblk(wpElems2, TPB), TPB, 0, stream>>>(W2d, w2dp, IN, HID, wpElems2);

  // ===== GATv2 layer 1 (IN->2x64) =====
  wmma_gemm_bias_h<128><<<rowTiles, (IN / 16) * 32, 0, stream>>>(xp, w1sp, b1s, A, N, IN);
  wmma_gemm_bias_h<128><<<rowTiles, (IN / 16) * 32, 0, stream>>>(xp, w1dp, b1d, B, N, IN);

  hipMemsetAsync(mm, 0, (size_t)N * 2 * 4, stream);
  edge_scores_k<<<nblk((long long)E * 2, TPB), TPB, 0, stream>>>(A, B, src, dst, attn1, C, mm, E, 2, HID);
  finalize_max_k<<<nblk(2 * N, TPB), TPB, 0, stream>>>(mm, m, 2 * N);
  hipMemsetAsync(s, 0, (size_t)N * 2 * 4, stream);
  edge_exp_k<<<nblk((long long)E * 2, TPB), TPB, 0, stream>>>(C, m, dst, s, E, 2);
  alpha_div_k<<<nblk((long long)E * 2, TPB), TPB, 0, stream>>>(C, s, dst, E, 2);
  hipMemsetAsync(B, 0, (size_t)N * IN * 4, stream);            // fd1 dead; B becomes g1 accum
  edge_aggregate_k<<<nblk((long long)E * IN, TPB), TPB, 0, stream>>>(A, C, src, dst, B, E, 2, HID);
  relu_k<<<nblk((long long)N * IN, TPB), TPB, 0, stream>>>(B, (long long)N * IN);

  // ===== GATv2 layer 2 (IN->64, 1 head) =====
  pack_a_k<<<nblk(apElems, TPB), TPB, 0, stream>>>(B, g1p, N, IN, apElems);
  wmma_gemm_bias_h<128><<<rowTiles, (HID / 16) * 32, 0, stream>>>(g1p, w2sp, b2s, fs2, N, HID);
  wmma_gemm_bias_h<128><<<rowTiles, (HID / 16) * 32, 0, stream>>>(g1p, w2dp, b2d, fd2, N, HID);

  hipMemsetAsync(mm, 0, (size_t)N * 4, stream);
  edge_scores_k<<<nblk(E, TPB), TPB, 0, stream>>>(fs2, fd2, src, dst, attn2, C, mm, E, 1, HID);
  finalize_max_k<<<nblk(N, TPB), TPB, 0, stream>>>(mm, m, N);
  hipMemsetAsync(s, 0, (size_t)N * 4, stream);
  edge_exp_k<<<nblk(E, TPB), TPB, 0, stream>>>(C, m, dst, s, E, 1);
  alpha_div_k<<<nblk(E, TPB), TPB, 0, stream>>>(C, s, dst, E, 1);
  hipMemsetAsync(g2, 0, (size_t)N * HID * 4, stream);          // g1(f32) dead; B[0:N*64] = g2 accum
  edge_aggregate_k<<<nblk((long long)E * HID, TPB), TPB, 0, stream>>>(fs2, C, src, dst, g2, E, 1, HID);
  relu_k<<<nblk((long long)N * HID, TPB), TPB, 0, stream>>>(g2, (long long)N * HID);

  // ===== GraphConv scores =====
  hipMemsetAsync(degO, 0, (size_t)N * 4, stream);
  hipMemsetAsync(degI, 0, (size_t)N * 4, stream);
  degrees_k<<<nblk(E, TPB), TPB, 0, stream>>>(src, dst, degO, degI, E);
  hproj_k<<<nblk(N, TPB), TPB, 0, stream>>>(g2, Wg, degO, hb, N, HID);
  hipMemsetAsync(agg, 0, (size_t)N * 4, stream);
  hagg_k<<<nblk(E, TPB), TPB, 0, stream>>>(hb, src, dst, agg, E);
  scores_k<<<nblk(N, TPB), TPB, 0, stream>>>(agg, degI, bg, sc, N);

  // ===== top-k gather + tanh gate =====
  topk_write_k<<<nblk(N, TPB), TPB, 0, stream>>>(sc, g2, (float*)d_out, N, k, HID);
}
